// MainNetwork_17592186044930
// MI455X (gfx1250) — matrix operand order repaired and checked by reading, not compiled, not run on hardware
//
#include <hip/hip_runtime.h>

// ---------------------------------------------------------------------------
// Problem constants (from reference): N_BR=8, BATCH=128, HID=512, ARR=T=256
// ---------------------------------------------------------------------------
constexpr int N_BR  = 8;
constexpr int BATCH = 128;
constexpr int HID   = 512;
constexpr int T     = 256;
constexpr int GATES = 4 * HID;          // 2048
constexpr int NT    = GATES / 16;       // 128 N-tiles of 16 gate columns
constexpr int KT    = HID / 32 + 1;     // 16 real K-tiles + 1 tile for (c_t, bias)
constexpr int FRAG_BYTES = 1024;        // one 32x16 bf16 B fragment (32 lanes * 32B)
constexpr size_t WSB_BYTES = (size_t)N_BR * NT * KT * FRAG_BYTES;  // ~17.8 MB

// LDS A-matrix: 16 rows x 544 bf16 cols (512 h + c_t + 1.0 + zeros), padded
constexpr int A_COLS     = 544;
constexpr int ROW_STRIDE = 1104;        // bytes; 1104/4=276 dwords, 276%64=20 -> conflict-free b128 rows

typedef __attribute__((ext_vector_type(16))) __bf16       v16bf;
typedef __attribute__((ext_vector_type(8)))  float        v8f;
typedef __attribute__((ext_vector_type(4)))  unsigned int u32x4;

union Frag { v16bf v; u32x4 u[2]; };

__device__ __forceinline__ float fast_rcp(float x)  { return __builtin_amdgcn_rcpf(x); }
__device__ __forceinline__ float fast_exp2(float x) { return __builtin_amdgcn_exp2f(x); }
__device__ __forceinline__ float sigmoid_f(float x) {
    return fast_rcp(1.0f + fast_exp2(-1.4426950408889634f * x));
}
__device__ __forceinline__ float tanh_f(float x) {
    // tanh(x) = 1 - 2/(exp(2x)+1) = 1 - 2/(2^(2x*log2e)+1)
    return 1.0f - 2.0f * fast_rcp(fast_exp2(2.8853900817779268f * x) + 1.0f);
}

// ---------------------------------------------------------------------------
// Prep: repack Whh (+ Wih/bias as extra K-tile) into bf16 B-fragment layout.
// B layout for v_wmma_f32_16x16x32_bf16 (32x16 K-major):
//   lane L (0..31) <-> N col L%16 ; VGPR j holds K=kl0 (lo16), K=kl0+1 (hi16) (table order)
// Fragment byte (lane,j,half) = lane*32 + j*4 + half*2.  One thread per dword.
// ---------------------------------------------------------------------------
__global__ void prep_kernel(const float* __restrict__ Whh,
                            const float* __restrict__ Wih,
                            const float* __restrict__ bih,
                            const float* __restrict__ bhh,
                            unsigned int* __restrict__ wsB) {
    long idx = (long)blockIdx.x * blockDim.x + threadIdx.x;
    long total = (long)N_BR * NT * KT * 256;   // dwords
    if (idx >= total) return;
    int  w    = (int)(idx & 255);
    long frag = idx >> 8;
    int lane = w >> 3, j = w & 7;
    int kt    = (int)(frag % KT);
    long tmp  = frag / KT;
    int ntile = (int)(tmp % NT);
    int br    = (int)(tmp / NT);
    int n0 = ntile * 16 + (lane & 15);
    int kl0 = (j < 4) ? (8 * (lane >> 4) + 2 * j) : (16 + 8 * (lane >> 4) + 2 * (j - 4));

    float v0, v1;
    if (kt < 16) {
        int k = kt * 32 + kl0;
        v0 = Whh[((long)br * GATES + n0) * HID + k];
        v1 = Whh[((long)br * GATES + n0) * HID + k + 1];
    } else if (kl0 == 0) {                      // K rows 512 (Wih column), 513 (total bias x ones)
        v0 = Wih[br * GATES + n0];
        v1 = bih[br * GATES + n0] + bhh[br * GATES + n0];
    } else {
        v0 = 0.0f; v1 = 0.0f;
    }
    __bf16 b0 = (__bf16)v0, b1 = (__bf16)v1;
    unsigned short u0, u1;
    __builtin_memcpy(&u0, &b0, 2);
    __builtin_memcpy(&u1, &b1, 2);
    wsB[idx] = (unsigned)u0 | ((unsigned)u1 << 16);
}

// ---------------------------------------------------------------------------
// Main: one WG per (branch, batch-chunk-of-16). 32 waves; wave w owns hidden
// cols [16w,16w+16): 4 gate tiles (i,f,g,o), 16x16 cell state in VGPRs.
// ---------------------------------------------------------------------------
__global__ __launch_bounds__(1024, 1)
void lstm_kernel(const float* __restrict__ x,      // (8,128,256)
                 const float* __restrict__ c,      // (128,256)
                 const float* __restrict__ hn0,    // (8,128,512)
                 const unsigned char* __restrict__ wsB,
                 float* __restrict__ r_ws) {       // (8,128,256)
    __shared__ __attribute__((aligned(16))) unsigned char lds_h[16 * ROW_STRIDE];
    __shared__ float lds_hcol[16];

    const int br  = blockIdx.x >> 3;
    const int b0  = (blockIdx.x & 7) * 16;
    const int tid = threadIdx.x;
    const int wave = tid >> 5, lane = tid & 31;

    // ---- init LDS A-matrix: h_{-1}=hn0 (bf16), col512=c[.,0], col513=1, rest 0
    for (int idx = tid; idx < 16 * A_COLS; idx += 1024) {
        int row = idx / A_COLS, col = idx % A_COLS;
        float v;
        if (col < HID)            v = hn0[((long)br * BATCH + b0 + row) * HID + col];
        else if (col == HID)      v = c[(b0 + row) * T + 0];
        else if (col == HID + 1)  v = 1.0f;
        else                      v = 0.0f;
        *(__bf16*)(lds_h + row * ROW_STRIDE + col * 2) = (__bf16)v;
    }
    __syncthreads();

    // B fragment base pointers: ntile = q*32 + wave  (q: 0=i,1=f,2=g,3=o)
    const unsigned char* bp0 = wsB + (((size_t)br * NT + ( 0 + wave)) * KT) * FRAG_BYTES + lane * 32;
    const unsigned char* bp1 = wsB + (((size_t)br * NT + (32 + wave)) * KT) * FRAG_BYTES + lane * 32;
    const unsigned char* bp2 = wsB + (((size_t)br * NT + (64 + wave)) * KT) * FRAG_BYTES + lane * 32;
    const unsigned char* bp3 = wsB + (((size_t)br * NT + (96 + wave)) * KT) * FRAG_BYTES + lane * 32;

    // A-fragment addressing (16-bit A 16x32): lanes 0-15 -> K {0..7,16..23},
    // lanes 16-31 -> K {8..15,24..31}; row M = lane&15.
    const int arow = lane & 15;
    const int asel = (lane >> 4) * 8;                 // element offset 0 or 8
    const unsigned char* abase = lds_h + arow * ROW_STRIDE;
    const int hrow = asel;                            // C-layout row base for this half-wave
    const int colbyte = (wave * 16 + (lane & 15)) * 2;

    v8f cc = (v8f)(0.0f);                             // cell state (reference: zeros)

    for (int t = 0; t < T; t++) {
        v8f acc_i = (v8f)(0.0f), acc_f = (v8f)(0.0f);
        v8f acc_g = (v8f)(0.0f), acc_o = (v8f)(0.0f);

        for (int kt = 0; kt < KT; kt++) {
            Frag a;
            const unsigned char* ap = abase + (kt * 32 + asel) * 2;
            a.u[0] = *(const u32x4*)(ap);
            a.u[1] = *(const u32x4*)(ap + 32);

            Frag b;
            const unsigned char* f0 = bp0 + kt * FRAG_BYTES;
            b.u[0] = *(const u32x4*)f0; b.u[1] = *(const u32x4*)(f0 + 16);
            acc_i = __builtin_amdgcn_wmma_f32_16x16x32_bf16(false, a.v, false, b.v, (short)0, acc_i, false, false);

            const unsigned char* f1 = bp1 + kt * FRAG_BYTES;
            b.u[0] = *(const u32x4*)f1; b.u[1] = *(const u32x4*)(f1 + 16);
            acc_f = __builtin_amdgcn_wmma_f32_16x16x32_bf16(false, a.v, false, b.v, (short)0, acc_f, false, false);

            const unsigned char* f2 = bp2 + kt * FRAG_BYTES;
            b.u[0] = *(const u32x4*)f2; b.u[1] = *(const u32x4*)(f2 + 16);
            acc_g = __builtin_amdgcn_wmma_f32_16x16x32_bf16(false, a.v, false, b.v, (short)0, acc_g, false, false);

            const unsigned char* f3 = bp3 + kt * FRAG_BYTES;
            b.u[0] = *(const u32x4*)f3; b.u[1] = *(const u32x4*)(f3 + 16);
            acc_o = __builtin_amdgcn_wmma_f32_16x16x32_bf16(false, a.v, false, b.v, (short)0, acc_o, false, false);
        }

        // ---- gate nonlinearities + state update (16x16 per wave, C-layout VGPRs)
        float hb[8];
        #pragma unroll
        for (int v = 0; v < 8; v++) {
            float ig = sigmoid_f(acc_i[v]);
            float fg = sigmoid_f(acc_f[v]);
            float gg = tanh_f(acc_g[v]);
            float og = sigmoid_f(acc_o[v]);
            float nc = fg * cc[v] + ig * gg;
            cc[v] = nc;
            hb[v] = og * tanh_f(nc);
        }

        __syncthreads();   // all reads of h_{t-1} done before overwrite

        #pragma unroll
        for (int v = 0; v < 8; v++) {   // C layout: VGPR v -> row v (lanes 0-15) / v+8 (16-31)
            *(__bf16*)(lds_h + (hrow + v) * ROW_STRIDE + colbyte) = (__bf16)hb[v];
        }
        if (wave == 31 && (lane & 15) == 15) {        // hidden col 511 -> f32 spill
            #pragma unroll
            for (int v = 0; v < 8; v++) lds_hcol[hrow + v] = hb[v];
        }
        if (wave == 0 && lane < 16 && t + 1 < T) {    // stage c_{t+1} into col 512
            *(__bf16*)(lds_h + lane * ROW_STRIDE + HID * 2) = (__bf16)c[(b0 + lane) * T + (t + 1)];
        }

        __syncthreads();   // h_t visible to all

        if (wave == 0 && lane < 16) {                 // r[br,b,t] = h_t[b,511]*x[br,b,t]
            long bi = (long)br * BATCH + b0 + lane;
            r_ws[bi * T + t] = lds_hcol[lane] * x[bi * T + t];
        }
    }
}

// ---------------------------------------------------------------------------
// Final: out[b,j] = bl[j] + sum_t (sum_br r[br,b,t]) * Wl[j,t]   (deterministic)
// ---------------------------------------------------------------------------
__global__ void final_kernel(const float* __restrict__ r_ws,
                             const float* __restrict__ Wl,
                             const float* __restrict__ bl,
                             float* __restrict__ out) {
    int idx = blockIdx.x * blockDim.x + threadIdx.x;
    if (idx >= BATCH * N_BR) return;
    int b = idx >> 3, j = idx & 7;
    float acc = bl[j];
    for (int t = 0; t < T; t++) {
        float r = 0.0f;
        for (int br = 0; br < N_BR; br++) r += r_ws[((long)br * BATCH + b) * T + t];
        acc += r * Wl[j * T + t];
    }
    out[idx] = acc;
}

extern "C" void kernel_launch(void* const* d_in, const int* in_sizes, int n_in,
                              void* d_out, int out_size, void* d_ws, size_t ws_size,
                              hipStream_t stream) {
    (void)in_sizes; (void)n_in; (void)out_size; (void)ws_size;
    const float* x    = (const float*)d_in[0];
    const float* c    = (const float*)d_in[1];
    const float* Wih  = (const float*)d_in[2];
    const float* Whh  = (const float*)d_in[3];
    const float* b_ih = (const float*)d_in[4];
    const float* b_hh = (const float*)d_in[5];
    const float* hn0  = (const float*)d_in[6];
    const float* Wl   = (const float*)d_in[7];
    const float* bl   = (const float*)d_in[8];

    unsigned char* wsB = (unsigned char*)d_ws;            // 17.8 MB bf16 B-fragments
    float* r_ws = (float*)(wsB + WSB_BYTES);              // 1 MB per-step outputs

    long prep_dwords = (long)N_BR * NT * KT * 256;
    prep_kernel<<<(unsigned)((prep_dwords + 255) / 256), 256, 0, stream>>>(
        Whh, Wih, b_ih, b_hh, (unsigned int*)wsB);

    lstm_kernel<<<N_BR * (BATCH / 16), 1024, 0, stream>>>(x, c, hn0, wsB, r_ws);

    final_kernel<<<(BATCH * N_BR + 255) / 256, 256, 0, stream>>>(r_ws, Wl, bl, (float*)d_out);
}
